// Sequence2_65094524338294
// MI455X (gfx1250) — matrix operand order repaired and checked by reading, not compiled, not run on hardware
//
#include <hip/hip_runtime.h>

// ---------------------------------------------------------------------------
// Seq2seq LSTM (T=64 enc + 64 dec, B=256, D=128, H=1024, L=2) for gfx1250.
// Strategy: bf16 WMMA (v_wmma_f32_16x16x32_bf16) for every GEMM, fp32 accum,
// fused gate-GEMM + LSTM pointwise per (step, layer). Weights converted to
// bf16 once per launch (~50MB -> L2 resident on MI455X's 192MB L2).
// ---------------------------------------------------------------------------

typedef float  v8f   __attribute__((ext_vector_type(8)));
typedef __bf16 v16bf __attribute__((ext_vector_type(16)));
typedef __bf16 v8bf  __attribute__((ext_vector_type(8)));

#define B_ 256
#define D_ 128
#define H_ 1024
#define T_ 64

union ABf { v16bf v; v8bf h[2]; };

__device__ __forceinline__ __bf16 f2bf(float f) {
  unsigned int u = __float_as_uint(f);
  unsigned int r = u + 0x7FFFu + ((u >> 16) & 1u);   // round-to-nearest-even
  unsigned short s = (unsigned short)(r >> 16);
  return __builtin_bit_cast(__bf16, s);
}

__device__ __forceinline__ float sigmoidf_(float x) {
  return 1.0f / (1.0f + __expf(-x));
}

// A fragment: 16x32 bf16, row m0+lane%16; lanes<16 hold K {0..7,16..23},
// lanes>=16 hold K {8..15,24..31}  (two 16B contiguous loads per lane).
__device__ __forceinline__ v16bf load_a_frag(const __bf16* __restrict__ A,
                                             int lda, int m0, int k0, int lane) {
  const int half = lane >> 4;
  const int m    = m0 + (lane & 15);
  const __bf16* p = A + (size_t)m * lda + k0 + half * 8;
  ABf u;
  u.h[0] = *reinterpret_cast<const v8bf*>(p);
  u.h[1] = *reinterpret_cast<const v8bf*>(p + 16);
  return u.v;
}

// B fragment: 32x16 bf16, col n0+lane%16; lanes<16 hold K {0..7,16..23},
// lanes>=16 hold K {8..15,24..31} (two 16B loads per lane from row-major W[N,K]).
__device__ __forceinline__ v16bf load_b_frag(const __bf16* __restrict__ W,
                                             int ldb, int n0, int k0, int lane) {
  const int half = lane >> 4;
  const int n    = n0 + (lane & 15);
  const __bf16* p = W + (size_t)n * ldb + k0 + half * 8;
  ABf u;
  u.h[0] = *reinterpret_cast<const v8bf*>(p);
  u.h[1] = *reinterpret_cast<const v8bf*>(p + 16);
  return u.v;
}

__device__ __forceinline__ v8f wmma_bf16(v16bf a, v16bf b, v8f c) {
  return __builtin_amdgcn_wmma_f32_16x16x32_bf16(false, a, false, b,
                                                 (short)0, c, false, false);
}

// ---------------------------------------------------------------------------
// Fused LSTM cell: gates = X·Wihᵀ + Hprev·Whhᵀ + bih + bhh ; pointwise update.
// One wave computes a 32(M)x16(N of H) tile of all four gates.
// Grid: 512 waves = 8 m-supertiles x 64 n-tiles -> 64 blocks x 256 threads.
// ---------------------------------------------------------------------------
__global__ __launch_bounds__(256) void lstm_cell_wmma(
    const __bf16* __restrict__ X,     // [B, Kx]
    const __bf16* __restrict__ Hp,    // [B, H]  h_{t-1}
    const __bf16* __restrict__ Wih,   // [4H, Kx]
    const __bf16* __restrict__ Whh,   // [4H, H]
    const float*  __restrict__ bih,   // [4H]
    const float*  __restrict__ bhh,   // [4H]
    const float*  __restrict__ Cprev, // [B, H]
    float*        __restrict__ Cnew,  // [B, H]
    float*        __restrict__ Hf,    // [B, H] fp32 h_t (for final outputs)
    __bf16*       __restrict__ Hb,    // [B, H] bf16 h_t (next-step GEMM A)
    int Kx)
{
  const int lane = threadIdx.x & 31;
  const int gw   = blockIdx.x * (blockDim.x >> 5) + (threadIdx.x >> 5);
  const int m0   = (gw >> 6) << 5;        // 8 supertiles of 32 rows
  const int n0   = (gw & 63) << 4;        // 64 tiles of 16 H-columns

  v8f acc[2][4] = {};                     // [m-subtile][gate]

  // ---- X · Wihᵀ contribution ----
  for (int k = 0; k < Kx; k += 32) {
    v16bf a0 = load_a_frag(X, Kx, m0,      k, lane);
    v16bf a1 = load_a_frag(X, Kx, m0 + 16, k, lane);
#pragma unroll
    for (int g = 0; g < 4; ++g) {
      v16bf b = load_b_frag(Wih, Kx, n0 + g * H_, k, lane);
      acc[0][g] = wmma_bf16(a0, b, acc[0][g]);
      acc[1][g] = wmma_bf16(a1, b, acc[1][g]);
    }
  }
  // ---- Hprev · Whhᵀ contribution ----
  for (int k = 0; k < H_; k += 32) {
    v16bf a0 = load_a_frag(Hp, H_, m0,      k, lane);
    v16bf a1 = load_a_frag(Hp, H_, m0 + 16, k, lane);
#pragma unroll
    for (int g = 0; g < 4; ++g) {
      v16bf b = load_b_frag(Whh, H_, n0 + g * H_, k, lane);
      acc[0][g] = wmma_bf16(a0, b, acc[0][g]);
      acc[1][g] = wmma_bf16(a1, b, acc[1][g]);
    }
  }

  // ---- fused pointwise LSTM update ----
  const int half = lane >> 4;
  const int n    = n0 + (lane & 15);
  const float bi = bih[n]          + bhh[n];
  const float bf = bih[H_ + n]     + bhh[H_ + n];
  const float bg = bih[2 * H_ + n] + bhh[2 * H_ + n];
  const float bo = bih[3 * H_ + n] + bhh[3 * H_ + n];

#pragma unroll
  for (int t = 0; t < 2; ++t) {
#pragma unroll
    for (int r = 0; r < 8; ++r) {
      const int m = m0 + t * 16 + r + half * 8;   // C/D layout: VGPR r -> M = r (+8 for upper lanes)
      const size_t idx = (size_t)m * H_ + n;
      float iv = sigmoidf_(acc[t][0][r] + bi);
      float fv = sigmoidf_(acc[t][1][r] + bf);
      float gv = tanhf(acc[t][2][r] + bg);
      float ov = sigmoidf_(acc[t][3][r] + bo);
      float c2 = fv * Cprev[idx] + iv * gv;
      float hn = ov * tanhf(c2);
      Cnew[idx] = c2;
      Hf[idx]   = hn;
      Hb[idx]   = f2bf(hn);
    }
  }
}

// ---------------------------------------------------------------------------
// FC: out[B,N] = A[B,K](bf16) · W[N,K]ᵀ + bias.  One 16x16 tile per wave.
// ---------------------------------------------------------------------------
__global__ __launch_bounds__(256) void fc_wmma(
    const __bf16* __restrict__ A,     // [B, K]
    const __bf16* __restrict__ W,     // [N, K]
    const float*  __restrict__ bias,  // [N]
    float*        __restrict__ outF,  // [B, N] or nullptr
    __bf16*       __restrict__ outB,  // [B, N] or nullptr
    int K, int N)
{
  const int lane   = threadIdx.x & 31;
  const int gw     = blockIdx.x * (blockDim.x >> 5) + (threadIdx.x >> 5);
  const int ntiles = N >> 4;
  const int m0     = (gw / ntiles) << 4;
  const int n0     = (gw % ntiles) << 4;

  v8f acc = {};
  for (int k = 0; k < K; k += 32) {
    v16bf a = load_a_frag(A, K, m0, k, lane);
    v16bf b = load_b_frag(W, K, n0, k, lane);
    acc = wmma_bf16(a, b, acc);
  }

  const int half = lane >> 4;
  const int n    = n0 + (lane & 15);
  const float bv = bias[n];
#pragma unroll
  for (int r = 0; r < 8; ++r) {
    const int m = m0 + r + half * 8;
    const float v = acc[r] + bv;
    if (outF) outF[(size_t)m * N + n] = v;
    if (outB) outB[(size_t)m * N + n] = f2bf(v);
  }
}

// ---------------------------------------------------------------------------
__global__ void f2bf_kernel(const float* __restrict__ in,
                            __bf16* __restrict__ out, int n) {
  for (int i = blockIdx.x * blockDim.x + threadIdx.x; i < n;
       i += gridDim.x * blockDim.x)
    out[i] = f2bf(in[i]);
}

// ---------------------------------------------------------------------------
extern "C" void kernel_launch(void* const* d_in, const int* in_sizes, int n_in,
                              void* d_out, int out_size, void* d_ws, size_t ws_size,
                              hipStream_t stream) {
  const float* input    = (const float*)d_in[0];
  const float* input_r  = (const float*)d_in[1];
  const float* h_enc0   = (const float*)d_in[2];
  const float* c_enc0   = (const float*)d_in[3];
  const float* encWih0  = (const float*)d_in[6];
  const float* encWhh0  = (const float*)d_in[7];
  const float* enc_bih0 = (const float*)d_in[8];
  const float* enc_bhh0 = (const float*)d_in[9];
  const float* encWih1  = (const float*)d_in[10];
  const float* encWhh1  = (const float*)d_in[11];
  const float* enc_bih1 = (const float*)d_in[12];
  const float* enc_bhh1 = (const float*)d_in[13];
  const float* decWih0  = (const float*)d_in[14];
  const float* decWhh0  = (const float*)d_in[15];
  const float* dec_bih0 = (const float*)d_in[16];
  const float* dec_bhh0 = (const float*)d_in[17];
  const float* decWih1  = (const float*)d_in[18];
  const float* decWhh1  = (const float*)d_in[19];
  const float* dec_bih1 = (const float*)d_in[20];
  const float* dec_bhh1 = (const float*)d_in[21];
  const float* fcEW     = (const float*)d_in[22];
  const float* fcEb     = (const float*)d_in[23];
  const float* fcDW     = (const float*)d_in[24];
  const float* fcDb     = (const float*)d_in[25];

  // ---- workspace bump allocator (256B aligned) ----
  char* wsb = (char*)d_ws;
  size_t off = 0;
  auto alloc = [&](size_t elems, size_t esz) -> void* {
    off = (off + 255) & ~(size_t)255;
    void* p = wsb + off;
    off += elems * esz;
    return p;
  };
  auto abf = [&](size_t n) { return (__bf16*)alloc(n, 2); };
  auto af  = [&](size_t n) { return (float*)alloc(n, 4); };

  const size_t BH = (size_t)B_ * H_;
  const size_t BD = (size_t)B_ * D_;

  __bf16* bWih0e = abf((size_t)4 * H_ * D_);
  __bf16* bWhh0e = abf((size_t)4 * H_ * H_);
  __bf16* bWih1e = abf((size_t)4 * H_ * H_);
  __bf16* bWhh1e = abf((size_t)4 * H_ * H_);
  __bf16* bWih0d = abf((size_t)4 * H_ * D_);
  __bf16* bWhh0d = abf((size_t)4 * H_ * H_);
  __bf16* bWih1d = abf((size_t)4 * H_ * H_);
  __bf16* bWhh1d = abf((size_t)4 * H_ * H_);
  __bf16* bFcE   = abf((size_t)D_ * H_);
  __bf16* bFcD   = abf((size_t)D_ * H_);
  __bf16* bXe    = abf((size_t)T_ * BD);
  __bf16* bXd    = abf((size_t)T_ * BD);
  __bf16* bOut0  = abf(BD);
  __bf16* hbf0[2] = {abf(BH), abf(BH)};
  __bf16* hbf1[2] = {abf(BH), abf(BH)};
  float*  hfp0 = af(BH);
  float*  hfp1 = af(BH);
  float*  cfp0[2] = {af(BH), af(BH)};
  float*  cfp1[2] = {af(BH), af(BH)};

  auto conv = [&](const float* s, __bf16* d, size_t n) {
    int blocks = (int)((n + 255) / 256);
    if (blocks > 1024) blocks = 1024;
    f2bf_kernel<<<blocks, 256, 0, stream>>>(s, d, (int)n);
  };

  // ---- one-time-per-launch conversions (deterministic) ----
  conv(encWih0, bWih0e, (size_t)4 * H_ * D_);
  conv(encWhh0, bWhh0e, (size_t)4 * H_ * H_);
  conv(encWih1, bWih1e, (size_t)4 * H_ * H_);
  conv(encWhh1, bWhh1e, (size_t)4 * H_ * H_);
  conv(decWih0, bWih0d, (size_t)4 * H_ * D_);
  conv(decWhh0, bWhh0d, (size_t)4 * H_ * H_);
  conv(decWih1, bWih1d, (size_t)4 * H_ * H_);
  conv(decWhh1, bWhh1d, (size_t)4 * H_ * H_);
  conv(fcEW, bFcE, (size_t)D_ * H_);
  conv(fcDW, bFcD, (size_t)D_ * H_);
  conv(input,   bXe, (size_t)T_ * BD);
  conv(input_r, bXd, (size_t)T_ * BD);
  // initial states (zeros in the reference, but be faithful)
  conv(h_enc0,      hbf0[0], BH);
  conv(h_enc0 + BH, hbf1[0], BH);
  hipMemcpyAsync(cfp0[0], c_enc0,      BH * 4, hipMemcpyDeviceToDevice, stream);
  hipMemcpyAsync(cfp1[0], c_enc0 + BH, BH * 4, hipMemcpyDeviceToDevice, stream);

  auto lstm = [&](const __bf16* X, int Kx, const __bf16* Hp,
                  const __bf16* Wih, const __bf16* Whh,
                  const float* bih, const float* bhh,
                  const float* Cp, float* Cn, float* Hf, __bf16* Hb) {
    lstm_cell_wmma<<<64, 256, 0, stream>>>(X, Hp, Wih, Whh, bih, bhh,
                                           Cp, Cn, Hf, Hb, Kx);
  };

  int p0 = 0, p1 = 0;

  // ---------------- encoder ----------------
  for (int t = 0; t < T_; ++t) {
    lstm(bXe + (size_t)t * BD, D_, hbf0[p0], bWih0e, bWhh0e, enc_bih0, enc_bhh0,
         cfp0[p0], cfp0[1 - p0], hfp0, hbf0[1 - p0]);
    p0 ^= 1;
    lstm(hbf0[p0], H_, hbf1[p1], bWih1e, bWhh1e, enc_bih1, enc_bhh1,
         cfp1[p1], cfp1[1 - p1], hfp1, hbf1[1 - p1]);
    p1 ^= 1;
  }

  float* out = (float*)d_out;
  const size_t OUT_HENC = (size_t)T_ * BD;          // after outputs [64,256,128]
  const size_t OUT_CENC = OUT_HENC + 2 * BH;
  const size_t OUT_HDEC = OUT_CENC + 2 * BH;
  const size_t OUT_CDEC = OUT_HDEC + 2 * BH;

  hipMemcpyAsync(out + OUT_HENC,      hfp0,     BH * 4, hipMemcpyDeviceToDevice, stream);
  hipMemcpyAsync(out + OUT_HENC + BH, hfp1,     BH * 4, hipMemcpyDeviceToDevice, stream);
  hipMemcpyAsync(out + OUT_CENC,      cfp0[p0], BH * 4, hipMemcpyDeviceToDevice, stream);
  hipMemcpyAsync(out + OUT_CENC + BH, cfp1[p1], BH * 4, hipMemcpyDeviceToDevice, stream);

  // fc_enc -> decoder step-0 input (bf16 only; fp32 copy not an output)
  fc_wmma<<<16, 256, 0, stream>>>(hbf1[p1], bFcE, fcEb,
                                  (float*)nullptr, bOut0, H_, D_);

  // ---------------- decoder (state continues from encoder) ----------------
  for (int t = 0; t < T_; ++t) {
    const __bf16* xt = (t == 0) ? bOut0 : bXd + (size_t)(t - 1) * BD;
    lstm(xt, D_, hbf0[p0], bWih0d, bWhh0d, dec_bih0, dec_bhh0,
         cfp0[p0], cfp0[1 - p0], hfp0, hbf0[1 - p0]);
    p0 ^= 1;
    lstm(hbf0[p0], H_, hbf1[p1], bWih1d, bWhh1d, dec_bih1, dec_bhh1,
         cfp1[p1], cfp1[1 - p1], hfp1, hbf1[1 - p1]);
    p1 ^= 1;
    fc_wmma<<<16, 256, 0, stream>>>(hbf1[p1], bFcD, fcDb,
                                    out + (size_t)t * BD, (__bf16*)nullptr, H_, D_);
  }

  hipMemcpyAsync(out + OUT_HDEC,      hfp0,     BH * 4, hipMemcpyDeviceToDevice, stream);
  hipMemcpyAsync(out + OUT_HDEC + BH, hfp1,     BH * 4, hipMemcpyDeviceToDevice, stream);
  hipMemcpyAsync(out + OUT_CDEC,      cfp0[p0], BH * 4, hipMemcpyDeviceToDevice, stream);
  hipMemcpyAsync(out + OUT_CDEC + BH, cfp1[p1], BH * 4, hipMemcpyDeviceToDevice, stream);
}
